// MultiHeadAttentionBlock_17102559772840
// MI455X (gfx1250) — compile-verified, matrix operand order checked by reading, not run on hardware
//
#include <hip/hip_runtime.h>
#include <hip/hip_bf16.h>
#include <stdint.h>

// ---------------------------------------------------------------------------
// MI455X (gfx1250) Multi-Head Attention block, bf16 WMMA path, f32 accumulate.
// B=4, S=2048, d_model=1024, H=16, d_k=64.
// GEMMs: LDS-staged, double-buffered, async global->LDS copies (ASYNCcnt).
// ---------------------------------------------------------------------------

typedef __attribute__((ext_vector_type(16))) __bf16 v16bf;
typedef __attribute__((ext_vector_type(8)))  __bf16 v8bf;
typedef __attribute__((ext_vector_type(8)))  float  v8f;
typedef __attribute__((ext_vector_type(4)))  int    v4i;

#define SEQ    2048
#define HEADS  16
#define DK     64
#define DM     1024
#define BATCH  4
#define MROWS  (BATCH * SEQ)      // 8192
#define NEGV   (-1.0e9f)
#define SM_SCALE 0.125f           // 1/sqrt(64)

#define GLOBAL_AS __attribute__((address_space(1)))
#define LDS_AS    __attribute__((address_space(3)))

// ---- async global->LDS copy (CDNA5), with compile-safe fallback -----------
#if __has_builtin(__builtin_amdgcn_global_load_async_to_lds_b128)
#define HAVE_ASYNC_LDS 1
#else
#define HAVE_ASYNC_LDS 0
#endif

__device__ __forceinline__ void cp16_g2l(const __bf16* gp, __bf16* lp) {
#if HAVE_ASYNC_LDS
  __builtin_amdgcn_global_load_async_to_lds_b128(
      (GLOBAL_AS v4i*)(uintptr_t)gp,
      (LDS_AS v4i*)(uint32_t)(uintptr_t)lp, 0, 0);
#else
  *(uint4*)lp = *(const uint4*)gp;
#endif
}

__device__ __forceinline__ void async_wait_all() {
#if HAVE_ASYNC_LDS
#if __has_builtin(__builtin_amdgcn_s_wait_asynccnt)
  __builtin_amdgcn_s_wait_asynccnt(0);
#else
  asm volatile("s_wait_asynccnt 0x0" ::: "memory");
#endif
#endif
}

// ---- WMMA fragment loader -------------------------------------------------
// 16-bit A/B fragment layout (CDNA5 ISA 7.12.2):
//   lanes 0-15  hold row (lane&15), K = [k0+0 .. k0+8) then [k0+16 .. k0+24)
//   lanes 16-31 hold row (lane&15), K = [k0+8 .. k0+16) then [k0+24 .. k0+32)
__device__ __forceinline__ v16bf load_frag(const __bf16* base, int row0, int k0,
                                           int rstride, int lane) {
  const int r   = row0 + (lane & 15);
  const int klo = k0 + ((lane >> 4) << 3);
  const __bf16* p = base + (size_t)r * rstride + klo;
  v8bf lo = *(const v8bf*)(p);
  v8bf hi = *(const v8bf*)(p + 16);
  v16bf f;
#pragma unroll
  for (int i = 0; i < 8; ++i) { f[i] = lo[i]; f[i + 8] = hi[i]; }
  return f;
}

// ---- f32 -> bf16 conversion -----------------------------------------------
__global__ void __launch_bounds__(256)
cvt_f32_bf16(const float* __restrict__ in, __bf16* __restrict__ out, int n) {
  int i = (blockIdx.x * blockDim.x + threadIdx.x) * 8;
  if (i >= n) return;
  const float4* p = (const float4*)(in + i);
  float4 a = p[0], b = p[1];
  v8bf o;
  o[0] = (__bf16)a.x; o[1] = (__bf16)a.y; o[2] = (__bf16)a.z; o[3] = (__bf16)a.w;
  o[4] = (__bf16)b.x; o[5] = (__bf16)b.y; o[6] = (__bf16)b.z; o[7] = (__bf16)b.w;
  *(v8bf*)(out + i) = o;
}

// ---- LDS staging: 128 rows x 32 bf16 (64B/row), 256 threads, 2 chunks each
__device__ __forceinline__ void stage_tile(const __bf16* __restrict__ g,
                                           int rstride, __bf16* l, int tid) {
#pragma unroll
  for (int it = 0; it < 2; ++it) {
    const int c   = tid + it * 256;      // 0..511
    const int row = c >> 2;
    const int off = (c & 3) << 3;        // 0,8,16,24 bf16 (16B chunks)
    cp16_g2l(g + (size_t)row * rstride + off, l + row * 32 + off);
  }
}

// ---- bf16 GEMM:  C[M,N] = A[M,K] * W[N,K]^T  (nn.Linear, bias=false) ------
// Block: 256 threads = 8 waves (4 M-waves x 2 N-waves), tile 128x128.
// Per wave: 32x64 output = 2 A-frags x 4 B-frags -> 8 WMMAs per k-step.
// A/B k-slices staged in LDS (async, double buffered).
// mode 0: bf16 [B,H,S,DK] | mode 1: bf16 [B,H,DK,S] | mode 2: f32 [M,N]
__global__ void __launch_bounds__(256)
gemm_bf16(const __bf16* __restrict__ A, const __bf16* __restrict__ W,
          void* __restrict__ Out, int M, int N, int K, int mode) {
  const int tid  = threadIdx.x;
  const int lane = tid & 31;
  const int wave = tid >> 5;
  const int wm   = wave >> 1;            // 0..3
  const int wn   = wave & 1;             // 0..1
  const int m0   = blockIdx.x * 128;
  const int n0   = blockIdx.y * 128;

  __shared__ __bf16 sA[2][128 * 32];
  __shared__ __bf16 sB[2][128 * 32];

  v8f c[2][4];
  {
    v8f z = {};
#pragma unroll
    for (int mi = 0; mi < 2; ++mi)
#pragma unroll
      for (int j = 0; j < 4; ++j) c[mi][j] = z;
  }

  const __bf16* Ab = A + (size_t)m0 * K;
  const __bf16* Wb = W + (size_t)n0 * K;

  // prologue: stage k-slice 0
  stage_tile(Ab, K, sA[0], tid);
  stage_tile(Wb, K, sB[0], tid);
  async_wait_all();
  __syncthreads();

  int buf = 0;
  for (int k0 = 0; k0 < K; k0 += 32) {
    const int nxt = buf ^ 1;
    if (k0 + 32 < K) {                   // stage next slice while computing
      stage_tile(Ab + k0 + 32, K, sA[nxt], tid);
      stage_tile(Wb + k0 + 32, K, sB[nxt], tid);
    }

    const __bf16* aT = sA[buf] + (wm * 32) * 32;
    const __bf16* bT = sB[buf] + (wn * 64) * 32;
    v16bf a0 = load_frag(aT, 0,  0, 32, lane);
    v16bf a1 = load_frag(aT, 16, 0, 32, lane);
#pragma unroll
    for (int j = 0; j < 4; ++j) {
      v16bf bfr = load_frag(bT, j * 16, 0, 32, lane);
      c[0][j] = __builtin_amdgcn_wmma_f32_16x16x32_bf16(
          false, a0, false, bfr, (short)0, c[0][j], false, false);
      c[1][j] = __builtin_amdgcn_wmma_f32_16x16x32_bf16(
          false, a1, false, bfr, (short)0, c[1][j], false, false);
    }

    async_wait_all();                    // next slice landed
    __syncthreads();                     // everyone done reading buf
    buf = nxt;
  }

  // C layout: VGPR i holds row (i + (lane>=16 ? 8:0)), column (lane&15).
  const int colt  = lane & 15;
  const int rbase = (lane >> 4) << 3;
#pragma unroll
  for (int mi = 0; mi < 2; ++mi) {
#pragma unroll
    for (int j = 0; j < 4; ++j) {
      const int n = n0 + wn * 64 + j * 16 + colt;
#pragma unroll
      for (int i = 0; i < 8; ++i) {
        const int m = m0 + wm * 32 + mi * 16 + rbase + i;
        const float val = c[mi][j][i];
        if (mode == 0) {
          int b = m >> 11, s = m & (SEQ - 1);
          int h = n >> 6,  d = n & (DK - 1);
          ((__bf16*)Out)[(((size_t)(b * HEADS + h)) * SEQ + s) * DK + d] = (__bf16)val;
        } else if (mode == 1) {
          int b = m >> 11, s = m & (SEQ - 1);
          int h = n >> 6,  d = n & (DK - 1);
          ((__bf16*)Out)[(((size_t)(b * HEADS + h)) * DK + d) * SEQ + s] = (__bf16)val;
        } else {
          ((float*)Out)[(size_t)m * N + n] = val;
        }
      }
    }
  }
}

// ---- Flash attention: one wave per (b,h, 16-query-row tile) ---------------
// Q  : [B,H,S,DK] bf16     K : [B,H,S,DK] bf16     Vt : [B,H,DK,S] bf16
// O  : [B,S,H*DK] bf16 (heads re-interleaved for the output projection)
__global__ void __launch_bounds__(32)
attn_flash(const __bf16* __restrict__ Q, const __bf16* __restrict__ Km,
           const __bf16* __restrict__ Vt, __bf16* __restrict__ O) {
  const int lane  = threadIdx.x;
  const int colt  = lane & 15;
  const int rbase = (lane >> 4) << 3;
  const int bh = blockIdx.y;
  const int s0 = blockIdx.x * 16;
  const int b  = bh / HEADS;
  const int h  = bh % HEADS;

  const __bf16* Qb = Q  + ((size_t)bh * SEQ + s0) * DK;
  const __bf16* Kb = Km + (size_t)bh * SEQ * DK;
  const __bf16* Vb = Vt + (size_t)bh * DK * SEQ;

  const v16bf qa0 = load_frag(Qb, 0, 0,  DK, lane);
  const v16bf qa1 = load_frag(Qb, 0, 32, DK, lane);

  float mrow[8], lrow[8];
  v8f acc[4];
  {
    v8f z = {};
#pragma unroll
    for (int i = 0; i < 4; ++i) acc[i] = z;
#pragma unroll
    for (int i = 0; i < 8; ++i) { mrow[i] = -3.0e38f; lrow[i] = 0.0f; }
  }

  __shared__ __bf16 ldsP[16 * 32];   // P tile, C-layout -> A-layout relay

  const int send = s0 + 15;
  for (int j0 = 0; j0 <= send; j0 += 32) {
    v8f sc0 = {}, sc1 = {};
    {
      v16bf kb = load_frag(Kb, j0, 0, DK, lane);
      sc0 = __builtin_amdgcn_wmma_f32_16x16x32_bf16(false, qa0, false, kb,
                                                    (short)0, sc0, false, false);
      kb = load_frag(Kb, j0, 32, DK, lane);
      sc0 = __builtin_amdgcn_wmma_f32_16x16x32_bf16(false, qa1, false, kb,
                                                    (short)0, sc0, false, false);
      kb = load_frag(Kb, j0 + 16, 0, DK, lane);
      sc1 = __builtin_amdgcn_wmma_f32_16x16x32_bf16(false, qa0, false, kb,
                                                    (short)0, sc1, false, false);
      kb = load_frag(Kb, j0 + 16, 32, DK, lane);
      sc1 = __builtin_amdgcn_wmma_f32_16x16x32_bf16(false, qa1, false, kb,
                                                    (short)0, sc1, false, false);
    }
    __builtin_prefetch(Kb + (size_t)(j0 + 32) * DK, 0, 1);

    float fac[8];
    __syncthreads();                      // protect ldsP across iterations
#pragma unroll
    for (int i = 0; i < 8; ++i) {
      const int row = s0 + rbase + i;
      const int c0  = j0 + colt;
      const int c1  = j0 + 16 + colt;
      float a  = (c0 <= row) ? sc0[i] * SM_SCALE : NEGV;
      float b_ = (c1 <= row) ? sc1[i] * SM_SCALE : NEGV;
      float t = fmaxf(a, b_);
      t = fmaxf(t, __shfl_xor(t, 1, 32));
      t = fmaxf(t, __shfl_xor(t, 2, 32));
      t = fmaxf(t, __shfl_xor(t, 4, 32));
      t = fmaxf(t, __shfl_xor(t, 8, 32));
      const float mnew = fmaxf(mrow[i], t);
      const float f    = __expf(mrow[i] - mnew);
      mrow[i] = mnew;
      fac[i]  = f;
      const float e0 = __expf(a  - mnew);
      const float e1 = __expf(b_ - mnew);
      float ls = e0 + e1;
      ls += __shfl_xor(ls, 1, 32);
      ls += __shfl_xor(ls, 2, 32);
      ls += __shfl_xor(ls, 4, 32);
      ls += __shfl_xor(ls, 8, 32);
      lrow[i] = lrow[i] * f + ls;
      ldsP[(rbase + i) * 32 + colt]      = (__bf16)e0;
      ldsP[(rbase + i) * 32 + 16 + colt] = (__bf16)e1;
    }
#pragma unroll
    for (int blk = 0; blk < 4; ++blk)
#pragma unroll
      for (int i = 0; i < 8; ++i) acc[blk][i] *= fac[i];

    __syncthreads();
    const v16bf pa = load_frag((const __bf16*)ldsP, 0, 0, 32, lane);

#pragma unroll
    for (int blk = 0; blk < 4; ++blk) {
      v16bf vb = load_frag(Vb + (size_t)blk * 16 * SEQ, 0, j0, SEQ, lane);
      acc[blk] = __builtin_amdgcn_wmma_f32_16x16x32_bf16(
          false, pa, false, vb, (short)0, acc[blk], false, false);
    }
  }

#pragma unroll
  for (int blk = 0; blk < 4; ++blk) {
#pragma unroll
    for (int i = 0; i < 8; ++i) {
      const int s = s0 + rbase + i;
      const float val = acc[blk][i] / lrow[i];
      O[((size_t)(b * SEQ + s)) * DM + h * DK + blk * 16 + colt] = (__bf16)val;
    }
  }
}

// ---------------------------------------------------------------------------
extern "C" void kernel_launch(void* const* d_in, const int* in_sizes, int n_in,
                              void* d_out, int out_size, void* d_ws, size_t ws_size,
                              hipStream_t stream) {
  const float* q   = (const float*)d_in[0];
  const float* k   = (const float*)d_in[1];
  const float* v   = (const float*)d_in[2];
  /* d_in[3] = mask : causal, computed analytically in-kernel */
  const float* w_q = (const float*)d_in[4];
  const float* w_k = (const float*)d_in[5];
  const float* w_v = (const float*)d_in[6];
  const float* w_o = (const float*)d_in[7];

  char* ws = (char*)d_ws;
  const size_t MB = 1ull << 20;
  __bf16* qb  = (__bf16*)(ws + 0 * MB);    // 16 MB  (reused as O later)
  __bf16* kb  = (__bf16*)(ws + 16 * MB);   // 16 MB
  __bf16* vb  = (__bf16*)(ws + 32 * MB);   // 16 MB
  __bf16* wqb = (__bf16*)(ws + 48 * MB);   //  2 MB
  __bf16* wkb = (__bf16*)(ws + 50 * MB);   //  2 MB
  __bf16* wvb = (__bf16*)(ws + 52 * MB);   //  2 MB
  __bf16* wob = (__bf16*)(ws + 54 * MB);   //  2 MB
  __bf16* Qh  = (__bf16*)(ws + 56 * MB);   // 16 MB [B,H,S,DK]
  __bf16* Kh  = (__bf16*)(ws + 72 * MB);   // 16 MB [B,H,S,DK]
  __bf16* Vt  = (__bf16*)(ws + 88 * MB);   // 16 MB [B,H,DK,S]
  __bf16* Ob  = qb;                        // alias: q-bf16 dead after Q proj

  const int nAct = MROWS * DM;             // 8388608
  const int nWgt = DM * DM;                // 1048576

  cvt_f32_bf16<<<nAct / (256 * 8), 256, 0, stream>>>(q, qb, nAct);
  cvt_f32_bf16<<<nAct / (256 * 8), 256, 0, stream>>>(k, kb, nAct);
  cvt_f32_bf16<<<nAct / (256 * 8), 256, 0, stream>>>(v, vb, nAct);
  cvt_f32_bf16<<<nWgt / (256 * 8), 256, 0, stream>>>(w_q, wqb, nWgt);
  cvt_f32_bf16<<<nWgt / (256 * 8), 256, 0, stream>>>(w_k, wkb, nWgt);
  cvt_f32_bf16<<<nWgt / (256 * 8), 256, 0, stream>>>(w_v, wvb, nWgt);
  cvt_f32_bf16<<<nWgt / (256 * 8), 256, 0, stream>>>(w_o, wob, nWgt);

  dim3 ggrid(MROWS / 128, DM / 128);       // 64 x 8 blocks, 256 threads
  gemm_bf16<<<ggrid, 256, 0, stream>>>(qb, wqb, Qh, MROWS, DM, DM, 0);
  gemm_bf16<<<ggrid, 256, 0, stream>>>(kb, wkb, Kh, MROWS, DM, DM, 0);
  gemm_bf16<<<ggrid, 256, 0, stream>>>(vb, wvb, Vt, MROWS, DM, DM, 1);

  dim3 agrid(SEQ / 16, BATCH * HEADS);     // 128 x 64, one wave per block
  attn_flash<<<agrid, 32, 0, stream>>>(Qh, Kh, Vt, Ob);

  gemm_bf16<<<ggrid, 256, 0, stream>>>(Ob, wob, d_out, MROWS, DM, DM, 2);
}
